// ISTFTHead_19481971655149
// MI455X (gfx1250) — matrix operand order repaired and checked by reading, not compiled, not run on hardware
//
#include <hip/hip_runtime.h>

// ---------------------------------------------------------------------------
// ISTFT head for MI455X (gfx1250, wave32, WMMA).
//   GEMM1 (bf16 WMMA, f32 acc): proj = x @ W^T + b, fused exp/clip/cos/sin
//          -> spec[T][1056] bf16  (cols 0..512 = a_k*re, 513..1025 = a_k*im)
//          register-blocked 2 M-tiles per wave (16 FLOP/B from cache).
//   GEMM2 (bf16 WMMA, f32 acc): frames = spec @ basis  (inverse rFFT as GEMM)
//          register-blocked 2 M x 4 N tiles per wave (~22 FLOP/B from cache).
//   OLA:   gather (<=4 frames per sample) + window-sum normalize.
// ---------------------------------------------------------------------------

typedef __attribute__((ext_vector_type(16))) __bf16 bf16x16;
typedef __attribute__((ext_vector_type(4)))  __bf16 bf16x4;
typedef __attribute__((ext_vector_type(8)))  float  f32x8;
typedef __attribute__((ext_vector_type(4)))  float  f32x4;

#define T_FRAMES 32768
#define DIM      512
#define N_FFT    1024
#define HOP      256
#define NBINS    513      // n_fft/2 + 1
#define OROWS    1026     // n_fft + 2
#define KPAD     1056     // 1026 padded up to multiple of 32
#define WROWS    1056     // W rows padded (phase B-tile can over-read to 1040)

union BFrag {
    bf16x16 v;
    f32x4   q[2];
};

// A-fragment: 16x32 bf16 tile, A row-major with leading dim ld.
// lane<16:  row=lane,    K = {0..7}  and {16..23}
// lane>=16: row=lane-16, K = {8..15} and {24..31}
__device__ __forceinline__ bf16x16 load_a_frag(const __bf16* __restrict__ A,
                                               int ld, int row0, int k0, int lane) {
    int m  = lane & 15;
    int kb = (lane >> 4) << 3;                 // 0 or 8
    const __bf16* p = A + (size_t)(row0 + m) * ld + k0 + kb;
    BFrag f;
    f.q[0] = *(const f32x4*)(p);               // K kb..kb+7
    f.q[1] = *(const f32x4*)(p + 16);          // K kb+16..kb+23
    return f.v;
}

// B-fragment: 32x16 bf16 tile of B, loaded from B^T stored row-major [N][K].
// lane holds K {kb..kb+7} and {kb+16..kb+23} of column n=lane&15, kb = 8*(lane>=16).
__device__ __forceinline__ bf16x16 load_b_frag(const __bf16* __restrict__ BT,
                                               int ld, int n0, int k0, int lane) {
    int n  = lane & 15;
    int kb = (lane >> 4) << 3;                 // 0 or 8
    const __bf16* p = BT + (size_t)(n0 + n) * ld + k0 + kb;
    BFrag f;
    f.q[0] = *(const f32x4*)(p);               // K kb..kb+7
    f.q[1] = *(const f32x4*)(p + 16);          // K kb+16..kb+23
    return f.v;
}

__device__ __forceinline__ f32x8 wmma_bf16(bf16x16 a, bf16x16 b, f32x8 c) {
    return __builtin_amdgcn_wmma_f32_16x16x32_bf16(
        /*neg_a=*/false, a, /*neg_b=*/false, b,
        /*c_mod=*/(short)0, c, /*reuse_a=*/false, /*reuse_b=*/false);
}

// ---------------------------- prep kernels ---------------------------------

// 4 elems/thread: 128-bit load, 64-bit store (x is 64 MB; keep it streaming).
__global__ __launch_bounds__(256) void k_cvt_x(const float* __restrict__ x,
                                               __bf16* __restrict__ xb, int n4) {
    int i = blockIdx.x * 256 + threadIdx.x;
    if (i >= n4) return;
    f32x4 v = *(const f32x4*)(x + (size_t)i * 4);
    bf16x4 o;
    o.x = (__bf16)v.x; o.y = (__bf16)v.y; o.z = (__bf16)v.z; o.w = (__bf16)v.w;
    *(bf16x4*)(xb + (size_t)i * 4) = o;
}

__global__ __launch_bounds__(256) void k_build_w(const float* __restrict__ W,
                                                 __bf16* __restrict__ Wb) {
    int i = blockIdx.x * 256 + threadIdx.x;
    if (i >= WROWS * DIM) return;
    int r = i >> 9;                            // / DIM
    Wb[i] = (r < OROWS) ? (__bf16)W[i] : (__bf16)0.0f;
}

// basisT[n][k], row-major [N_FFT][KPAD]:
//   k <  513 : cos(2*pi*k*n/1024)
//   k < 1026 : -sin(2*pi*(k-513)*n/1024)
//   else     : 0
__global__ __launch_bounds__(256) void k_build_basis(__bf16* __restrict__ bT) {
    int i = blockIdx.x * 256 + threadIdx.x;
    if (i >= N_FFT * KPAD) return;
    int nrow = i / KPAD;
    int k    = i - nrow * KPAD;
    float v = 0.0f;
    const float w = 6.28318530717958647f * (1.0f / 1024.0f);
    if (k < NBINS) {
        int r = (k * nrow) & (N_FFT - 1);      // exact periodic reduction
        v = __cosf(w * (float)r);
    } else if (k < OROWS) {
        int r = ((k - NBINS) * nrow) & (N_FFT - 1);
        v = -__sinf(w * (float)r);
    }
    bT[i] = (__bf16)v;
}

__global__ __launch_bounds__(256) void k_zero_spec_pad(__bf16* __restrict__ spec) {
    int i = blockIdx.x * 256 + threadIdx.x;
    const int padc = KPAD - OROWS;             // 30
    if (i >= T_FRAMES * padc) return;
    int t = i / padc;
    int c = OROWS + (i - t * padc);
    spec[(size_t)t * KPAD + c] = (__bf16)0.0f;
}

// ------------------------- GEMM1 + activation ------------------------------
// Per wave: 2 M-tiles x (one 16x16 mag tile + paired phase tile at +513).
// 1024 row-groups x 33 bin-tiles = 33792 waves = 4224 blocks of 8 waves.

__global__ __launch_bounds__(256) void k_gemm1_spec(const __bf16* __restrict__ Xb,
                                                    const __bf16* __restrict__ Wb,
                                                    const float*  __restrict__ bias,
                                                    __bf16* __restrict__ spec) {
    int lane = threadIdx.x & 31;
    int wid  = blockIdx.x * 8 + (threadIdx.x >> 5);
    int mg    = wid / 33;                      // row-group (32 rows)
    int ntile = wid - mg * 33;
    int row0  = mg * 32;
    int cb    = ntile * 16;                    // bin base (0..512 step 16)

    f32x8 cm0 = {}, cm1 = {}, cp0 = {}, cp1 = {};
    #pragma unroll 4
    for (int k0 = 0; k0 < DIM; k0 += 32) {
        bf16x16 a0 = load_a_frag(Xb, DIM, row0,      k0, lane);
        bf16x16 a1 = load_a_frag(Xb, DIM, row0 + 16, k0, lane);
        bf16x16 bm = load_b_frag(Wb, DIM, cb,         k0, lane);   // mag rows
        bf16x16 bp = load_b_frag(Wb, DIM, cb + NBINS, k0, lane);   // phase rows
        cm0 = wmma_bf16(a0, bm, cm0);
        cm1 = wmma_bf16(a1, bm, cm1);
        cp0 = wmma_bf16(a0, bp, cp0);
        cp1 = wmma_bf16(a1, bp, cp1);
    }

    int n = lane & 15;
    int k = cb + n;
    if (k >= NBINS) return;                    // last tile: only col 512 valid
    int mb = (lane >> 4) << 3;
    float bmag = bias[k];
    float bph  = bias[NBINS + k];
    float ak = (k == 0 || k == N_FFT / 2) ? (1.0f / N_FFT) : (2.0f / N_FFT);
    #pragma unroll
    for (int v = 0; v < 8; ++v) {
        {
            int t = row0 + mb + v;
            float mag = fminf(__expf(cm0[v] + bmag), 100.0f);
            float ph  = cp0[v] + bph;
            spec[(size_t)t * KPAD + k]         = (__bf16)(mag * __cosf(ph) * ak);
            spec[(size_t)t * KPAD + NBINS + k] = (__bf16)(mag * __sinf(ph) * ak);
        }
        {
            int t = row0 + 16 + mb + v;
            float mag = fminf(__expf(cm1[v] + bmag), 100.0f);
            float ph  = cp1[v] + bph;
            spec[(size_t)t * KPAD + k]         = (__bf16)(mag * __cosf(ph) * ak);
            spec[(size_t)t * KPAD + NBINS + k] = (__bf16)(mag * __sinf(ph) * ak);
        }
    }
}

// ----------------------------- GEMM2 (iDFT) --------------------------------
// Per wave: 32 rows x 64 cols (2 M-tiles x 4 N-tiles):
//   A fragments reused 4x, B fragments reused 2x -> ~22 FLOP/B from cache.
// 1024 row-groups x 16 col-groups = 16384 waves = 2048 blocks.

__global__ __launch_bounds__(256) void k_gemm2_frames(const __bf16* __restrict__ spec,
                                                      const __bf16* __restrict__ basisT,
                                                      float* __restrict__ frames) {
    int lane = threadIdx.x & 31;
    int wid  = blockIdx.x * 8 + (threadIdx.x >> 5);
    int mg   = wid >> 4;                       // row-group (32 rows)
    int ng   = wid & 15;                       // col-group (64 cols)
    int row0 = mg * 32;
    int ncol = ng * 64;

    f32x8 c00 = {}, c01 = {}, c02 = {}, c03 = {};
    f32x8 c10 = {}, c11 = {}, c12 = {}, c13 = {};
    for (int k0 = 0; k0 < KPAD; k0 += 32) {
        bf16x16 a0 = load_a_frag(spec, KPAD, row0,      k0, lane);
        bf16x16 a1 = load_a_frag(spec, KPAD, row0 + 16, k0, lane);
        bf16x16 b0 = load_b_frag(basisT, KPAD, ncol,      k0, lane);
        bf16x16 b1 = load_b_frag(basisT, KPAD, ncol + 16, k0, lane);
        bf16x16 b2 = load_b_frag(basisT, KPAD, ncol + 32, k0, lane);
        bf16x16 b3 = load_b_frag(basisT, KPAD, ncol + 48, k0, lane);
        c00 = wmma_bf16(a0, b0, c00);
        c10 = wmma_bf16(a1, b0, c10);
        c01 = wmma_bf16(a0, b1, c01);
        c11 = wmma_bf16(a1, b1, c11);
        c02 = wmma_bf16(a0, b2, c02);
        c12 = wmma_bf16(a1, b2, c12);
        c03 = wmma_bf16(a0, b3, c03);
        c13 = wmma_bf16(a1, b3, c13);
    }

    int n  = lane & 15;
    int mb = (lane >> 4) << 3;
    #pragma unroll
    for (int v = 0; v < 8; ++v) {
        size_t r0 = (size_t)(row0 + mb + v) * N_FFT + ncol + n;
        frames[r0]      = c00[v];
        frames[r0 + 16] = c01[v];
        frames[r0 + 32] = c02[v];
        frames[r0 + 48] = c03[v];
        size_t r1 = r0 + (size_t)16 * N_FFT;
        frames[r1]      = c10[v];
        frames[r1 + 16] = c11[v];
        frames[r1 + 32] = c12[v];
        frames[r1 + 48] = c13[v];
    }
}

// ----------------------- overlap-add + normalize ---------------------------
// Gather form: each output sample reads the <=4 frames that cover it.

__global__ __launch_bounds__(256) void k_ola(const float* __restrict__ frames,
                                             const float* __restrict__ window,
                                             float* __restrict__ audio, int t_len) {
    int s = blockIdx.x * 256 + threadIdx.x;
    if (s >= t_len) return;
    int ihi = s >> 8;                          // floor(s/HOP)
    if (ihi > T_FRAMES - 1) ihi = T_FRAMES - 1;
    int lo = s - (N_FFT - HOP);                // s - 768 -> ceil((s-1023)/256)
    int ilo = (lo > 0) ? (lo >> 8) : 0;
    float num = 0.0f, den = 0.0f;
    for (int i = ilo; i <= ihi; ++i) {
        int j = s - (i << 8);
        float w = window[j];
        num += frames[(size_t)i * N_FFT + j] * w;
        den += w;
    }
    audio[s] = (den != 0.0f) ? (num / den) : num;
}

// ---------------------------------------------------------------------------

extern "C" void kernel_launch(void* const* d_in, const int* in_sizes, int n_in,
                              void* d_out, int out_size, void* d_ws, size_t ws_size,
                              hipStream_t stream) {
    const float* x      = (const float*)d_in[0];   // [1, 32768, 512]
    const float* W      = (const float*)d_in[1];   // [1026, 512]
    const float* bias   = (const float*)d_in[2];   // [1026]
    const float* window = (const float*)d_in[3];   // [1024]
    float* audio = (float*)d_out;

    // workspace layout (all offsets 256B-aligned)
    char* ws = (char*)d_ws;
    const size_t SZ_XB    = (size_t)T_FRAMES * DIM * 2;          //  33,554,432
    const size_t SZ_WB    = (size_t)WROWS * DIM * 2;             //   1,081,344
    const size_t SZ_BASIS = (size_t)N_FFT * KPAD * 2;            //   2,162,688
    const size_t SZ_SPEC  = (size_t)T_FRAMES * KPAD * 2;         //  69,206,016
    __bf16* Xb     = (__bf16*)(ws);
    __bf16* Wb     = (__bf16*)(ws + SZ_XB);
    __bf16* basisT = (__bf16*)(ws + SZ_XB + SZ_WB);
    __bf16* spec   = (__bf16*)(ws + SZ_XB + SZ_WB + SZ_BASIS);
    float*  frames = (float*)(ws + SZ_XB + SZ_WB + SZ_BASIS + SZ_SPEC);

    // prep
    {
        int n4 = (T_FRAMES * DIM) / 4;
        k_cvt_x<<<(n4 + 255) / 256, 256, 0, stream>>>(x, Xb, n4);
    }
    k_build_w<<<(WROWS * DIM + 255) / 256, 256, 0, stream>>>(W, Wb);
    k_build_basis<<<(N_FFT * KPAD + 255) / 256, 256, 0, stream>>>(basisT);
    k_zero_spec_pad<<<(T_FRAMES * (KPAD - OROWS) + 255) / 256, 256, 0, stream>>>(spec);

    // GEMM1 + activation -> spec
    {
        int waves = (T_FRAMES / 32) * 33;      // 33792
        k_gemm1_spec<<<waves / 8, 256, 0, stream>>>(Xb, Wb, bias, spec);
    }

    // GEMM2 (inverse rFFT as GEMM) -> frames
    {
        int waves = (T_FRAMES / 32) * 16;      // 16384
        k_gemm2_frames<<<waves / 8, 256, 0, stream>>>(spec, basisT, frames);
    }

    // overlap-add + window normalize -> audio
    k_ola<<<(out_size + 255) / 256, 256, 0, stream>>>(frames, window, audio, out_size);
}
